// HeadPac_506806141152
// MI455X (gfx1250) — matrix operand order repaired and checked by reading, not compiled, not run on hardware
//
#include <hip/hip_runtime.h>

// ---------------------------------------------------------------------------
// PAC-VGG forward for MI455X (gfx1250): implicit-GEMM 3x3 PAC convs on WMMA.
// K-order = ij*Cin + c (Cin pow2) so each aligned 32-chunk has a uniform tap.
// Per wave: 64 Couts x 32 pixels (8 accumulators). Per K-chunk: 2 B gathers +
// 4 A fragments (2x ds_load_b128 each) feed 8 v_wmma_f32_16x16x32_f16.
// ---------------------------------------------------------------------------

typedef __attribute__((ext_vector_type(16))) _Float16 v16h;
typedef __attribute__((ext_vector_type(8)))  _Float16 v8h;
typedef __attribute__((ext_vector_type(8)))  float    v8f;

#define KBLK 288   // LDS K-block: 64 rows * 288 * 2B = 36 KB; divides 576/1152/2304

// ---------------------------------------------------------------------------
// 2x2 maxpool over [2,3,Hin,Win] -> [2,3,Hin/2,Win/2]
// ---------------------------------------------------------------------------
__global__ void maxpool2_k(const float* __restrict__ in, float* __restrict__ out,
                           int Hin, int Win) {
  const int Ho = Hin >> 1, Wo = Win >> 1;
  const int total = 2 * 3 * Ho * Wo;
  int idx = blockIdx.x * blockDim.x + threadIdx.x;
  if (idx >= total) return;
  const int wo = idx % Wo;
  int t = idx / Wo;
  const int ho = t % Ho;
  const int bc = t / Ho;
  const float* p = in + ((size_t)bc * Hin + 2 * ho) * Win + 2 * wo;
  out[idx] = fmaxf(fmaxf(p[0], p[1]), fmaxf(p[Win], p[Win + 1]));
}

// ---------------------------------------------------------------------------
// Gaussian PAC kernel: k[b,ij,h,w] = exp(-0.5*sum_c (g_patch - g_center)^2)
// ---------------------------------------------------------------------------
__global__ void pac_kernel_k(const float* __restrict__ g, float* __restrict__ kout,
                             int H, int W) {
  const int total = 2 * H * W;
  int idx = blockIdx.x * blockDim.x + threadIdx.x;
  if (idx >= total) return;
  const size_t hw = (size_t)H * W;
  const int b = idx / (H * W);
  const int p = idx % (H * W);
  const int h = p / W, w = p % W;
  const float* gb = g + (size_t)b * 3 * hw;
  const float c0 = gb[p], c1 = gb[hw + p], c2 = gb[2 * hw + p];
  float* ko = kout + (size_t)b * 9 * hw + p;
#pragma unroll
  for (int ij = 0; ij < 9; ++ij) {
    const int i = ij / 3, j = ij % 3;
    const int hh = h + i - 1, ww = w + j - 1;
    float g0 = 0.f, g1 = 0.f, g2 = 0.f;
    if (hh >= 0 && hh < H && ww >= 0 && ww < W) {
      const float* q = gb + (size_t)hh * W + ww;
      g0 = q[0]; g1 = q[hw]; g2 = q[2 * hw];
    }
    const float d0 = g0 - c0, d1 = g1 - c1, d2 = g2 - c2;
    ko[(size_t)ij * hw] = expf(-0.5f * (d0 * d0 + d1 * d1 + d2 * d2));
  }
}

// ---------------------------------------------------------------------------
// PAC conv on v_wmma_f32_16x16x32_f16.
// grid = (pixels/128, Cout/64, B=2), block = 128 (4 waves).
// Wave w: pixels [base, base+32), Cout rows [coutBase, coutBase+64).
// ---------------------------------------------------------------------------
__global__ __launch_bounds__(128) void pac_conv_wmma(
    const float* __restrict__ x, const float* __restrict__ kpac,
    const float* __restrict__ wgt, float* __restrict__ out,
    int Cin, int log2Cin, int Cout, int Hin, int Win,
    int Ho, int Wo, int log2Wo, int stride, int relu_out) {
  __shared__ _Float16 lds_w[64 * KBLK];

  const int tid    = threadIdx.x;
  const int wave   = tid >> 5;
  const int lane   = tid & 31;
  const int hiHalf = lane >> 4;   // lane group 16..31
  const int mn     = lane & 15;   // M within sub-tile / N (pixel) column
  const int b      = blockIdx.z;
  const int coutBase = blockIdx.y * 64;
  const int K = 9 << log2Cin;     // Cin * 9

  const size_t hwin = (size_t)Hin * Win;
  const float* xb = x + (size_t)b * Cin * hwin;
  const float* kb = kpac + (size_t)b * 9 * hwin;

  // Two pixel groups of 16 per wave.
  const int pbase = (blockIdx.x * 4 + wave) * 32;
  int   ph[2], pw[2];
  float kfs[2][9];
  int   offs[2][9];
#pragma unroll
  for (int png = 0; png < 2; ++png) {
    const int p = pbase + png * 16 + mn;
    ph[png] = p >> log2Wo;
    pw[png] = p & (Wo - 1);
    const int xh0 = ph[png] * stride - 1;
    const int xw0 = pw[png] * stride - 1;
    const float* kp = kb + (size_t)(ph[png] * stride) * Win
                         + (size_t)(pw[png] * stride);
#pragma unroll
    for (int ij = 0; ij < 9; ++ij) {
      const int hh = xh0 + ij / 3;
      const int ww = xw0 + ij % 3;
      const bool inb = (hh >= 0) & (hh < Hin) & (ww >= 0) & (ww < Win);
      offs[png][ij] = inb ? hh * Win + ww : 0;
      kfs[png][ij]  = inb ? kp[(size_t)ij * hwin] : 0.f;
    }
  }

  v8f acc[2][4] = {};

  for (int kblock = 0; kblock < K; kblock += KBLK) {
    const int krem  = K - kblock;
    const int kcur  = krem < KBLK ? krem : KBLK;
    const int kload = (kcur + 31) & ~31;

    __syncthreads();
    // Stage the 64-row weight slice for this K-block into LDS (f32 -> f16).
    for (int m = 0; m < 64; ++m) {
      const float* wrow = wgt + (size_t)(coutBase + m) * K;  // [Cout][Cin*9]
      for (int k = tid; k < kload; k += 128) {
        const int kg = kblock + k;
        float v = 0.f;
        if (kg < K) {
          const int c  = kg & (Cin - 1);
          const int ij = kg >> log2Cin;
          v = wrow[c * 9 + ij];
        }
        lds_w[m * KBLK + k] = (_Float16)v;
      }
    }
    __syncthreads();

    // Pull the next weight K-block toward L2 while we compute on this one.
    if (kblock + KBLK < K)
      __builtin_prefetch(wgt + (size_t)(coutBase + (tid >> 1)) * K
                             + kblock + KBLK + (tid & 1) * 16, 0, 1);

    for (int kk = 0; kk < kload; kk += 32) {
      // ---- Two B fragments: contiguous k order, same as A (lane<16: K 0..15; else 16..31).
      v16h bf[2];
      if (Cin == 1) {
        // Generic path (only the first conv, K=9): per-element tap decode.
        const int kgBase = kblock + kk + (hiHalf << 4);
#pragma unroll
        for (int png = 0; png < 2; ++png) {
#pragma unroll
          for (int e = 0; e < 16; ++e) {
            const int kg = kgBase + e;
            float v = 0.f;
            if (kg < K) v = xb[offs[png][kg]] * kfs[png][kg];  // ij == kg, c == 0
            bf[png][e] = (_Float16)v;
          }
        }
      } else {
        // Fast path: aligned 32-chunk lies within one tap ij (Cin >= 32).
        const int kg0 = kblock + kk;
        const int ij  = kg0 >> log2Cin;
        const int cB  = (kg0 & (Cin - 1)) + (hiHalf << 4);
        const float* bpc = xb + (size_t)cB * hwin;
#pragma unroll
        for (int png = 0; png < 2; ++png) {
          const float kfv = kfs[png][ij];
          const float* bp = bpc + offs[png][ij];
#pragma unroll
          for (int e = 0; e < 16; ++e)
            bf[png][e] = (_Float16)(bp[(size_t)e * hwin] * kfv);
        }
      }

      // ---- 4 A fragments from LDS; each is two contiguous 8-half runs
      // (16-bit 16x32 A layout) -> two 16B LDS reads per fragment.
#pragma unroll
      for (int m4 = 0; m4 < 4; ++m4) {
        const _Float16* arow =
            &lds_w[(m4 * 16 + mn) * KBLK + kk + (hiHalf << 4)];
        const v8h alo = *(const v8h*)arow;          // K = base+0 .. base+7
        const v8h ahi = *(const v8h*)(arow + 8);    // K = base+8 .. base+15
        v16h af;
#pragma unroll
        for (int e = 0; e < 8; ++e) { af[e] = alo[e]; af[e + 8] = ahi[e]; }

#pragma unroll
        for (int png = 0; png < 2; ++png)
          acc[png][m4] = __builtin_amdgcn_wmma_f32_16x16x32_f16(
              false, af, false, bf[png], (short)0, acc[png][m4], false, false);
      }
    }
  }

  // D epilogue, 32-bit C/D layout: VGPR r -> M = r + 8*hiHalf, N = lane&15.
#pragma unroll
  for (int png = 0; png < 2; ++png) {
#pragma unroll
    for (int m4 = 0; m4 < 4; ++m4) {
#pragma unroll
      for (int r = 0; r < 8; ++r) {
        const int m = m4 * 16 + r + (hiHalf << 3);
        float v = acc[png][m4][r];
        if (relu_out) v = fmaxf(v, 0.f);
        out[((size_t)(b * Cout + coutBase + m) * Ho + ph[png]) * Wo + pw[png]] = v;
      }
    }
  }
}

// ---------------------------------------------------------------------------
// Host side
// ---------------------------------------------------------------------------
static void launch_conv(const float* x, const float* kpac, const float* w,
                        float* out, int Cin, int Cout, int Hin, int Win,
                        int stride, int relu, hipStream_t s) {
  const int Ho = Hin / stride, Wo = Win / stride;
  const int log2Cin = __builtin_ctz((unsigned)Cin);
  const int log2Wo  = __builtin_ctz((unsigned)Wo);
  const int pixels = Ho * Wo;                  // multiple of 128 for all layers
  dim3 grid(pixels / 128, Cout / 64, 2);
  pac_conv_wmma<<<grid, 128, 0, s>>>(x, kpac, w, out, Cin, log2Cin, Cout,
                                     Hin, Win, Ho, Wo, log2Wo, stride, relu);
}

extern "C" void kernel_launch(void* const* d_in, const int* in_sizes, int n_in,
                              void* d_out, int out_size, void* d_ws, size_t ws_size,
                              hipStream_t stream) {
  (void)in_sizes; (void)n_in; (void)out_size; (void)ws_size;
  const float* x     = (const float*)d_in[0];
  const float* guide = (const float*)d_in[1];
  const float* W[14];
  for (int i = 0; i < 14; ++i) W[i] = (const float*)d_in[2 + i];

  float* out = (float*)d_out;
  // Output layout (flat, return order): h1..h5 then k1..k5.
  float* h1 = out;                      // 2*64*256*256  = 8388608
  float* h2 = out + 8388608;            // 2*64*128*128  = 2097152
  float* h3 = out + 10485760;           // 2*128*64*64   = 1048576
  float* h4 = out + 11534336;           // 2*128*32*32   =  262144
  float* h5 = out + 11796480;           // 2*256*16*16   =  131072
  float* k1 = out + 11927552;           // 2*9*256*256   = 1179648
  float* k2 = out + 13107200;           // 2*9*128*128   =  294912
  float* k3 = out + 13402112;           // 2*9*64*64     =   73728
  float* k4 = out + 13475840;           // 2*9*32*32     =   18432
  float* k5 = out + 13494272;           // 2*9*16*16     =    4608

  float* tA = (float*)d_ws;             // ping-pong activation buffers
  float* tB = tA + 8388608;
  float* g2 = tB + 8388608;             // 2*3*128*128 = 98304
  float* g3 = g2 + 98304;               // 2*3*64*64   = 24576
  float* g4 = g3 + 24576;               // 2*3*32*32   =  6144
  float* g5 = g4 + 6144;                // 2*3*16*16   =  1536

  // Guide pyramid.
  maxpool2_k<<<(2 * 3 * 128 * 128 + 255) / 256, 256, 0, stream>>>(guide, g2, 256, 256);
  maxpool2_k<<<(2 * 3 * 64 * 64 + 255) / 256, 256, 0, stream>>>(g2, g3, 128, 128);
  maxpool2_k<<<(2 * 3 * 32 * 32 + 255) / 256, 256, 0, stream>>>(g3, g4, 64, 64);
  maxpool2_k<<<(2 * 3 * 16 * 16 + 255) / 256, 256, 0, stream>>>(g4, g5, 32, 32);

  // PAC kernels (written straight into their output slices, reused by convs).
  pac_kernel_k<<<(2 * 256 * 256 + 255) / 256, 256, 0, stream>>>(guide, k1, 256, 256);
  pac_kernel_k<<<(2 * 128 * 128 + 255) / 256, 256, 0, stream>>>(g2, k2, 128, 128);
  pac_kernel_k<<<(2 * 64 * 64 + 255) / 256, 256, 0, stream>>>(g3, k3, 64, 64);
  pac_kernel_k<<<(2 * 32 * 32 + 255) / 256, 256, 0, stream>>>(g4, k4, 32, 32);
  pac_kernel_k<<<(2 * 16 * 16 + 255) / 256, 256, 0, stream>>>(g5, k5, 16, 16);

  // Level 1 (256x256). Intermediates stored post-ReLU so inputs are ready.
  launch_conv(x,  k1, W[0], tA, 1,  64, 256, 256, 1, 1, stream);
  launch_conv(tA, k1, W[1], h1, 64, 64, 256, 256, 1, 1, stream);
  launch_conv(h1, k1, W[2], tA, 64, 64, 256, 256, 2, 1, stream);
  // Level 2 (128x128).
  launch_conv(tA, k2, W[3], tB, 64, 64, 128, 128, 1, 1, stream);
  launch_conv(tB, k2, W[4], h2, 64, 64, 128, 128, 1, 1, stream);
  launch_conv(h2, k2, W[5], tA, 64, 64, 128, 128, 2, 1, stream);
  // Level 3 (64x64).
  launch_conv(tA, k3, W[6], tB, 64,  128, 64, 64, 1, 1, stream);
  launch_conv(tB, k3, W[7], h3, 128, 128, 64, 64, 1, 1, stream);
  launch_conv(h3, k3, W[8], tA, 128, 128, 64, 64, 2, 1, stream);
  // Level 4 (32x32).
  launch_conv(tA, k4, W[9],  tB, 128, 128, 32, 32, 1, 1, stream);
  launch_conv(tB, k4, W[10], h4, 128, 128, 32, 32, 1, 1, stream);
  launch_conv(h4, k4, W[11], tA, 128, 128, 32, 32, 2, 1, stream);
  // Level 5 (16x16); h5 has no trailing ReLU.
  launch_conv(tA, k5, W[12], tB, 128, 256, 16, 16, 1, 1, stream);
  launch_conv(tB, k5, W[13], h5, 256, 256, 16, 16, 1, 0, stream);
}
